// CNNTransformer_37503654429228
// MI455X (gfx1250) — matrix operand order repaired and checked by reading, not compiled, not run on hardware
//
#include <hip/hip_runtime.h>
#include <hip/hip_bf16.h>

typedef __attribute__((ext_vector_type(16))) _Float16 v16h;
typedef __attribute__((ext_vector_type(8)))  _Float16 v8h;
typedef __attribute__((ext_vector_type(8)))  float    v8f;

#define DEV __device__ __forceinline__

// Padded tile geometry: 16x16 interior inside an 18x18 zero-halo frame.
#define PT    18
#define PTILE 324     // 18*18
// Padded full image: 256x256 interior inside 258x258.
#define PIMG  258

// ---------------------------------------------------------------------------
// WMMA fragment loaders for V_WMMA_F32_16X16X32_F16 (CDNA5 wave32 layouts).
// A (16x32, row-major source, K contiguous):
//   lanes 0-15  : row M=lane,    elems 0-7 = K k0..k0+7,   elems 8-15 = K k0+16..k0+23
//   lanes 16-31 : row M=lane-16, elems 0-7 = K k0+8..k0+15, elems 8-15 = K k0+24..k0+31
DEV v16h make_a_frag(const _Float16* __restrict__ row, int k0) {
  const int hi = (threadIdx.x & 31) >> 4;
  const _Float16* p = row + k0 + hi * 8;
  v8h lo = *(const v8h*)(p);
  v8h hh = *(const v8h*)(p + 16);
  v16h a;
#pragma unroll
  for (int i = 0; i < 8; ++i) { a[i] = lo[i]; a[i + 8] = hh[i]; }
  return a;
}

// B (32x16): lane holds one column, same K order as A (lanes 0-15: K0-7,16-23;
// lanes 16-31: K8-15,24-31). colbase points at (column, K=0) of a K-contiguous column.
DEV v16h load_b_frag(const _Float16* __restrict__ colbase) {
  const int hi = (threadIdx.x & 31) >> 4;
  const _Float16* p = colbase + hi * 8;
  v8h lo = *(const v8h*)(p);
  v8h hh = *(const v8h*)(p + 16);
  v16h b;
#pragma unroll
  for (int i = 0; i < 8; ++i) { b[i] = lo[i]; b[i + 8] = hh[i]; }
  return b;
}

DEV v8f wmma_f16(v16h a, v16h b, v8f c) {
  return __builtin_amdgcn_wmma_f32_16x16x32_f16(false, a, false, b, (short)0, c,
                                                false, false);
}

// ---------------------------------------------------------------------------
// Zero-fill for halo'd f16 buffers (8 halves per thread).
__global__ void zero_h_kernel(_Float16* __restrict__ p, size_t n8) {
  const size_t i = (size_t)blockIdx.x * 256 + threadIdx.x;
  if (i < n8) {
    v8h z = {};
    *(v8h*)(p + i * 8) = z;
  }
}

// ---------------------------------------------------------------------------
// Weight reorder: (Cout, Cin, KH, KW) fp32 -> f16 [co][r*Cin + ci], scaled.
__global__ void reorder_w_kernel(const float* __restrict__ win,
                                 _Float16* __restrict__ wout,
                                 int Cout, int Cin, int KHW, float scale) {
  const int idx = blockIdx.x * 256 + threadIdx.x;
  const int total = Cout * Cin * KHW;
  if (idx >= total) return;
  const int co  = idx / (Cin * KHW);
  const int rem = idx - co * (Cin * KHW);
  const int ci  = rem / KHW;
  const int r   = rem - ci * KHW;
  wout[(size_t)co * (Cin * KHW) + (size_t)r * Cin + ci] =
      (_Float16)(win[idx] * scale);
}

// ---------------------------------------------------------------------------
// sem conv (1x1, 3->64) + relu + tile rearrange + positional encoding.
// act  : fp32 [b][tile][c][p]   actH : f16 [b][tile][PTILE][c] (zero halo)
__global__ void sem_pos_kernel(const float* __restrict__ x,
                               const float* __restrict__ pos,
                               const float* __restrict__ sw,
                               const float* __restrict__ sb,
                               float* __restrict__ act,
                               _Float16* __restrict__ actH, int b) {
  const int p  = blockIdx.x * 256 + threadIdx.x;  // 0..65535
  const int y  = p >> 8, xq = p & 255;
  const float x0 = x[((size_t)b * 3 + 0) * 65536 + p];
  const float x1 = x[((size_t)b * 3 + 1) * 65536 + p];
  const float x2 = x[((size_t)b * 3 + 2) * 65536 + p];
  const int ty = y >> 4, py = y & 15, tx = xq >> 4, px = xq & 15;
  const int tile = ty * 16 + tx, pl = py * 16 + px;
  const int ppad = (py + 1) * PT + (px + 1);
  const size_t fbase = ((size_t)b * 256 + tile) * 16384;
  const size_t hbase = ((size_t)b * 256 + tile) * (PTILE * 64);
  for (int c = 0; c < 64; ++c) {
    float v = sb[c] + sw[c * 3 + 0] * x0 + sw[c * 3 + 1] * x1 + sw[c * 3 + 2] * x2;
    v = fmaxf(v, 0.f);
    v += pos[(size_t)tile * 64 + c];
    act[fbase + (size_t)c * 256 + pl]        = v;
    actH[hbase + (size_t)ppad * 64 + c]      = (_Float16)v;
  }
}

// ---------------------------------------------------------------------------
// Per-tile 3x3 conv (pad 1) as implicit WMMA GEMM on halo-padded input.
//   inH : f16 [256 tiles][PTILE][CIN]  (zero halo -> no predication)
//   W   : f16 [Mout][9*CIN]            (k = tap*CIN + ci)
// block = 128 (4 waves, one pixel-block each), grid = (4, Mout/16, 256).
#define OUT_QK 0  // outH f16 [tile][m*256 + p]          (Q/K per-head activations)
#define OUT_V  1  // outH f16 [(m*256 + p)*256 + tile]   (V transposed for PV GEMM)
#define OUT_U  2  // outF f32 [tile][Mout][p]
template <int CIN, int MODE>
__global__ void conv3x3_tile_wmma(const _Float16* __restrict__ inH,
                                  const _Float16* __restrict__ W,
                                  int Mout,
                                  float* __restrict__ outF,
                                  _Float16* __restrict__ outH) {
  const int lane = threadIdx.x & 31;
  const int wave = threadIdx.x >> 5;
  const int col  = lane & 15;
  const int hi   = lane >> 4;
  const int nblk = blockIdx.x * 4 + wave;
  const int mblk = blockIdx.y;
  const int tile = blockIdx.z;
  const int p  = nblk * 16 + col;
  const int py = p >> 4, px = p & 15;
  const _Float16* ctr  = inH + (size_t)tile * (PTILE * CIN) +
                         (size_t)((py + 1) * PT + (px + 1)) * CIN;
  const _Float16* Arow = W + (size_t)(mblk * 16 + col) * (9 * CIN);
  v8f acc = {};
#pragma unroll
  for (int r = 0; r < 9; ++r) {
    const int doff = ((r / 3) - 1) * PT + ((r % 3) - 1);
    const _Float16* bcol = ctr + (ptrdiff_t)doff * CIN;
#pragma unroll
    for (int kc = 0; kc < CIN; kc += 32) {
      v16h b = load_b_frag(bcol + kc);
      v16h a = make_a_frag(Arow, r * CIN + kc);
      acc = wmma_f16(a, b, acc);
    }
  }
  const int m0 = mblk * 16 + hi * 8;
  if (MODE == OUT_U) {
#pragma unroll
    for (int i = 0; i < 8; ++i)
      outF[((size_t)tile * Mout + (m0 + i)) * 256 + p] = acc[i];
  } else if (MODE == OUT_QK) {
#pragma unroll
    for (int i = 0; i < 8; ++i)
      outH[(size_t)tile * 16384 + (size_t)(m0 + i) * 256 + p] = (_Float16)acc[i];
  } else {  // OUT_V
#pragma unroll
    for (int i = 0; i < 8; ++i)
      outH[((size_t)(m0 + i) * 256 + p) * 256 + tile] = (_Float16)acc[i];
  }
}

// ---------------------------------------------------------------------------
// Attention scores for one (b, head): S[j][q] = sum_f qh[j][f] * kh[q][f].
// qh/kh: f16 [256][16384]; K loop = 512 WMMA chunks. grid = (4, 16), block 128.
__global__ void attn_s_wmma(const _Float16* __restrict__ qh,
                            const _Float16* __restrict__ kh,
                            float* __restrict__ S) {
  const int lane = threadIdx.x & 31;
  const int wave = threadIdx.x >> 5;
  const int col = lane & 15, hi = lane >> 4;
  const int qblk = blockIdx.x * 4 + wave;
  const int jblk = blockIdx.y;
  const _Float16* arow = qh + (size_t)(jblk * 16 + col) * 16384;
  const _Float16* brow = kh + (size_t)(qblk * 16 + col) * 16384;
  v8f acc = {};
  for (int k = 0; k < 16384; k += 32) {
    __builtin_prefetch(brow + k + 1024, 0, 1);  // global_prefetch ahead in K
    v16h a = make_a_frag(arow, k);
    v16h b = load_b_frag(brow + k);
    acc = wmma_f16(a, b, acc);
  }
  const int j0 = jblk * 16 + hi * 8;
  const int q  = qblk * 16 + col;
#pragma unroll
  for (int i = 0; i < 8; ++i) S[(size_t)(j0 + i) * 256 + q] = acc[i];
}

// Row softmax over 256 entries; one wave32 per row, 8 elems/lane.
__global__ void softmax256(const float* __restrict__ S, _Float16* __restrict__ P) {
  const int row = blockIdx.x;
  const int lane = threadIdx.x;
  float v[8];
  float mx = -3.0e38f;
#pragma unroll
  for (int i = 0; i < 8; ++i) {
    v[i] = S[(size_t)row * 256 + i * 32 + lane];
    mx = fmaxf(mx, v[i]);
  }
#pragma unroll
  for (int off = 16; off > 0; off >>= 1) mx = fmaxf(mx, __shfl_xor(mx, off, 32));
  float s = 0.f;
#pragma unroll
  for (int i = 0; i < 8; ++i) { v[i] = __expf(v[i] - mx); s += v[i]; }
#pragma unroll
  for (int off = 16; off > 0; off >>= 1) s += __shfl_xor(s, off, 32);
  const float inv = 1.f / s;
#pragma unroll
  for (int i = 0; i < 8; ++i)
    P[(size_t)row * 256 + i * 32 + lane] = (_Float16)(v[i] * inv);
}

// O[j][f] = sum_q P[j][q] * V[q][f]; V stored transposed f16 [f][q].
// Output scattered into halo-padded U-conv layout oh f16 [tile][PTILE][512].
__global__ void attn_pv_wmma(const _Float16* __restrict__ P,
                             const _Float16* __restrict__ vh,
                             _Float16* __restrict__ oh, int head) {
  const int lane = threadIdx.x & 31;
  const int wave = threadIdx.x >> 5;
  const int col = lane & 15, hi = lane >> 4;
  const int fblk = blockIdx.x * 4 + wave;  // 1024 feature blocks
  const int jblk = blockIdx.y;             // 16 tile blocks
  const _Float16* arow = P + (size_t)(jblk * 16 + col) * 256;
  const _Float16* brow = vh + (size_t)(fblk * 16 + col) * 256;
  v8f acc = {};
#pragma unroll
  for (int k = 0; k < 256; k += 32) {
    v16h a = make_a_frag(arow, k);
    v16h b = load_b_frag(brow + k);
    acc = wmma_f16(a, b, acc);
  }
  const int j0 = jblk * 16 + hi * 8;
  const int f  = fblk * 16 + col;
  const int c = f >> 8, p = f & 255;
  const int ppad = ((p >> 4) + 1) * PT + (p & 15) + 1;
#pragma unroll
  for (int i = 0; i < 8; ++i)
    oh[((size_t)(j0 + i) * PTILE + ppad) * 512 + head * 64 + c] =
        (_Float16)acc[i];
}

// ---------------------------------------------------------------------------
// Plain WMMA GEMM for the 1x1-conv MLP: out[m][n] = sum_k A[m][k] * Bm[n][k].
#define GEMM_MLP1 0  // +bias, relu, outH f16 [n][256] (m contiguous)
#define GEMM_MLP2 1  // +bias, outF f32 [tile][64][p] with n = tile*256+p
template <int K, int MODE>
__global__ void gemm_wmma(const _Float16* __restrict__ A,
                          const _Float16* __restrict__ Bm,
                          const float* __restrict__ bias,
                          float* __restrict__ outF, _Float16* __restrict__ outH) {
  const int lane = threadIdx.x & 31;
  const int wave = threadIdx.x >> 5;
  const int col = lane & 15, hi = lane >> 4;
  const int nblk = blockIdx.x * 4 + wave;
  const int mblk = blockIdx.y;
  const _Float16* arow = A + (size_t)(mblk * 16 + col) * K;
  const _Float16* brow = Bm + (size_t)(nblk * 16 + col) * K;
  v8f acc = {};
#pragma unroll
  for (int k = 0; k < K; k += 32) {
    v16h a = make_a_frag(arow, k);
    v16h b = load_b_frag(brow + k);
    acc = wmma_f16(a, b, acc);
  }
  const int m0 = mblk * 16 + hi * 8;
  const int n  = nblk * 16 + col;
  if (MODE == GEMM_MLP1) {
    v8h r;
#pragma unroll
    for (int i = 0; i < 8; ++i)
      r[i] = (_Float16)fmaxf(acc[i] + bias[m0 + i], 0.f);
    *(v8h*)(outH + (size_t)n * 256 + m0) = r;
  } else {
    const int tile = n >> 8, p = n & 255;
#pragma unroll
    for (int i = 0; i < 8; ++i)
      outF[((size_t)tile * 64 + (m0 + i)) * 256 + p] = acc[i] + bias[m0 + i];
  }
}

// ---------------------------------------------------------------------------
// Residual add + per-tile LayerNorm over 16384 elems, elementwise affine.
// a,res: fp32 [tile][64][256]. Writes fp32 (same layout), f16 channel-last
// (padded tile layout when outHPadded, else dense [tile][256][64]), and
// optionally the halo-padded full-image f16 layout for the output convs.
__global__ void residual_ln(const float* __restrict__ a,
                            const float* __restrict__ res,
                            const float* __restrict__ lw,
                            const float* __restrict__ lb,
                            float* __restrict__ outF,
                            _Float16* __restrict__ outH, int outHPadded,
                            _Float16* __restrict__ imgH) {
  const int tile = blockIdx.x, tid = threadIdx.x;
  __shared__ float sred[16];
  __shared__ float sMu, sRstd;
  const size_t base = (size_t)tile * 16384;
  float s = 0.f, ss = 0.f;
  for (int i = tid; i < 16384; i += 256) {
    const float v = a[base + i] + res[base + i];
    s += v; ss += v * v;
  }
#pragma unroll
  for (int off = 16; off > 0; off >>= 1) {
    s  += __shfl_xor(s, off, 32);
    ss += __shfl_xor(ss, off, 32);
  }
  if ((tid & 31) == 0) { sred[tid >> 5] = s; sred[8 + (tid >> 5)] = ss; }
  __syncthreads();
  if (tid == 0) {
    float S = 0.f, SS = 0.f;
    for (int k2 = 0; k2 < 8; ++k2) { S += sred[k2]; SS += sred[8 + k2]; }
    const float mu = S * (1.f / 16384.f);
    const float var = SS * (1.f / 16384.f) - mu * mu;
    sMu = mu; sRstd = rsqrtf(var + 1e-5f);
  }
  __syncthreads();
  const float mu = sMu, rstd = sRstd;
  const int ty = tile >> 4, tx = tile & 15;
  for (int i = tid; i < 16384; i += 256) {
    const float v = a[base + i] + res[base + i];
    const float y = (v - mu) * rstd * lw[i] + lb[i];
    outF[base + i] = y;
    const int c = i >> 8, p = i & 255;
    const int py = p >> 4, px = p & 15;
    if (outHPadded) {
      const int ppad = (py + 1) * PT + (px + 1);
      outH[((size_t)tile * PTILE + ppad) * 64 + c] = (_Float16)y;
    } else {
      outH[((size_t)tile * 256 + p) * 64 + c] = (_Float16)y;
    }
    if (imgH) {
      const int gy = ty * 16 + py + 1, gx = tx * 16 + px + 1;
      imgH[((size_t)gy * PIMG + gx) * 64 + c] = (_Float16)y;
    }
  }
}

// ---------------------------------------------------------------------------
// Full-image 3x3 conv (pad 1), 64->64, +bias, relu.
// imgH: halo-padded f16 [PIMG][PIMG][64]; outH dense f16 [y][x][64].
__global__ void conv3x3_img_wmma(const _Float16* __restrict__ imgH,
                                 const _Float16* __restrict__ W,  // [64][576]
                                 const float* __restrict__ bias,
                                 _Float16* __restrict__ outH) {
  const int lane = threadIdx.x & 31;
  const int wave = threadIdx.x >> 5;
  const int col = lane & 15, hi = lane >> 4;
  const int nblk = blockIdx.x * 4 + wave;  // 4096 pixel blocks
  const int mblk = blockIdx.y;             // 4
  const int p  = nblk * 16 + col;
  const int py = p >> 8, px = p & 255;
  const _Float16* ctr  = imgH + (size_t)((py + 1) * PIMG + (px + 1)) * 64;
  const _Float16* Arow = W + (size_t)(mblk * 16 + col) * 576;
  v8f acc = {};
#pragma unroll
  for (int r = 0; r < 9; ++r) {
    const int doff = ((r / 3) - 1) * PIMG + ((r % 3) - 1);
    const _Float16* bcol = ctr + (ptrdiff_t)doff * 64;
#pragma unroll
    for (int kc = 0; kc < 64; kc += 32) {
      v16h b = load_b_frag(bcol + kc);
      v16h a = make_a_frag(Arow, r * 64 + kc);
      acc = wmma_f16(a, b, acc);
    }
  }
  const int m0 = mblk * 16 + hi * 8;
  v8h rv;
#pragma unroll
  for (int i = 0; i < 8; ++i)
    rv[i] = (_Float16)fmaxf(acc[i] + bias[m0 + i], 0.f);
  *(v8h*)(outH + (size_t)p * 64 + m0) = rv;
}

// Final 1x1 conv 64->3 (+bias) into d_out fp32 (B,3,256,256).
__global__ void out2_kernel(const _Float16* __restrict__ h,
                            const float* __restrict__ w2,
                            const float* __restrict__ b2,
                            float* __restrict__ out, int b) {
  const int p = blockIdx.x * 256 + threadIdx.x;
  float a0 = b2[0], a1 = b2[1], a2 = b2[2];
  const _Float16* hp = h + (size_t)p * 64;
  for (int c = 0; c < 64; ++c) {
    const float hv = (float)hp[c];
    a0 += w2[c] * hv;
    a1 += w2[64 + c] * hv;
    a2 += w2[128 + c] * hv;
  }
  out[((size_t)b * 3 + 0) * 65536 + p] = a0;
  out[((size_t)b * 3 + 1) * 65536 + p] = a1;
  out[((size_t)b * 3 + 2) * 65536 + p] = a2;
}

// ---------------------------------------------------------------------------
extern "C" void kernel_launch(void* const* d_in, const int* in_sizes, int n_in,
                              void* d_out, int out_size, void* d_ws,
                              size_t ws_size, hipStream_t stream) {
  (void)in_sizes; (void)n_in; (void)out_size; (void)ws_size;
  const float* x     = (const float*)d_in[0];
  const float* pos   = (const float*)d_in[1];
  const float* sem_w = (const float*)d_in[2];
  const float* sem_b = (const float*)d_in[3];
  const float* qw    = (const float*)d_in[4];
  const float* kw    = (const float*)d_in[5];
  const float* vw    = (const float*)d_in[6];
  const float* uw    = (const float*)d_in[7];
  const float* w1    = (const float*)d_in[8];
  const float* b1    = (const float*)d_in[9];
  const float* w2    = (const float*)d_in[10];
  const float* b2    = (const float*)d_in[11];
  const float* ln1w  = (const float*)d_in[12];
  const float* ln1b  = (const float*)d_in[13];
  const float* ln2w  = (const float*)d_in[14];
  const float* ln2b  = (const float*)d_in[15];
  const float* ow1   = (const float*)d_in[16];
  const float* ob1   = (const float*)d_in[17];
  const float* ow2   = (const float*)d_in[18];
  const float* ob2   = (const float*)d_in[19];
  float* out = (float*)d_out;

  char* wsb = (char*)d_ws;
  size_t off = 0;
  auto alloc = [&](size_t bytes) -> char* {
    char* p = wsb + off;
    off += (bytes + 255) & ~(size_t)255;
    return p;
  };
  const size_t actH_n = (size_t)2 * 256 * PTILE * 64;   // halves
  const size_t oh_n   = (size_t)256 * PTILE * 512;      // halves (per b)
  const size_t img_n  = (size_t)2 * PIMG * PIMG * 64;   // halves (both b)

  float*    act    = (float*)   alloc((size_t)2 * 256 * 16384 * 4);  // t (fp32)
  _Float16* actH   = (_Float16*)alloc(actH_n * 2);   // halo-padded conv input
  _Float16* qh     = (_Float16*)alloc((size_t)256 * 16384 * 2);  // per (b,head)
  _Float16* kh     = (_Float16*)alloc((size_t)256 * 16384 * 2);
  _Float16* vh     = (_Float16*)alloc((size_t)256 * 16384 * 2);  // transposed
  float*    Smat   = (float*)   alloc((size_t)256 * 256 * 4);
  _Float16* Pmat   = (_Float16*)alloc((size_t)256 * 256 * 2);
  _Float16* oh     = (_Float16*)alloc(oh_n * 2);     // halo-padded, per b
  float*    tmp    = (float*)   alloc((size_t)256 * 16384 * 4);  // per b
  float*    ident  = (float*)   alloc((size_t)256 * 16384 * 4);  // per b
  _Float16* identH = (_Float16*)alloc((size_t)256 * 16384 * 2);  // dense, per b
  _Float16* m1     = (_Float16*)alloc((size_t)65536 * 256 * 2);  // per b
  _Float16* imgH   = (_Float16*)alloc(img_n * 2);    // halo-padded, both b
  _Float16* h1H    = (_Float16*)alloc((size_t)65536 * 64 * 2);   // per b
  _Float16* wqre   = (_Float16*)alloc((size_t)512 * 576 * 2);
  _Float16* wkre   = (_Float16*)alloc((size_t)512 * 576 * 2);
  _Float16* wvre   = (_Float16*)alloc((size_t)512 * 576 * 2);
  _Float16* wure   = (_Float16*)alloc((size_t)64 * 4608 * 2);
  _Float16* w1re   = (_Float16*)alloc((size_t)256 * 64 * 2);
  _Float16* w2re   = (_Float16*)alloc((size_t)64 * 256 * 2);
  _Float16* wo1re  = (_Float16*)alloc((size_t)64 * 576 * 2);

  // Zero the halo'd buffers every call (interiors are rewritten; halos must
  // read as zero for the padding-free conv loops). 8 halves per thread.
  auto zero_h = [&](_Float16* p, size_t n) {
    const size_t n8 = n / 8;
    zero_h_kernel<<<(unsigned)((n8 + 255) / 256), 256, 0, stream>>>(p, n8);
  };
  zero_h(actH, actH_n);
  zero_h(oh, oh_n);
  zero_h(imgH, img_n);

  // Stage 0: semantic 1x1 conv + relu + tiling + positional encoding.
  for (int b = 0; b < 2; ++b)
    sem_pos_kernel<<<256, 256, 0, stream>>>(x, pos, sem_w, sem_b, act, actH, b);

  const float qk_scale = 0.25f;  // 1 / (16*16)^0.25, folded into Q and K weights

  for (int l = 0; l < 2; ++l) {
    reorder_w_kernel<<<(512 * 576 + 255) / 256, 256, 0, stream>>>(
        qw + (size_t)l * 512 * 64 * 9, wqre, 512, 64, 9, qk_scale);
    reorder_w_kernel<<<(512 * 576 + 255) / 256, 256, 0, stream>>>(
        kw + (size_t)l * 512 * 64 * 9, wkre, 512, 64, 9, qk_scale);
    reorder_w_kernel<<<(512 * 576 + 255) / 256, 256, 0, stream>>>(
        vw + (size_t)l * 512 * 64 * 9, wvre, 512, 64, 9, 1.f);
    reorder_w_kernel<<<(64 * 4608 + 255) / 256, 256, 0, stream>>>(
        uw + (size_t)l * 64 * 512 * 9, wure, 64, 512, 9, 1.f);
    reorder_w_kernel<<<(256 * 64 + 255) / 256, 256, 0, stream>>>(
        w1 + (size_t)l * 256 * 64, w1re, 256, 64, 1, 1.f);
    reorder_w_kernel<<<(64 * 256 + 255) / 256, 256, 0, stream>>>(
        w2 + (size_t)l * 64 * 256, w2re, 64, 256, 1, 1.f);

    for (int b = 0; b < 2; ++b) {
      const _Float16* aH = actH + (size_t)b * 256 * (PTILE * 64);
      float* aF = act + (size_t)b * 256 * 16384;

      for (int h = 0; h < 8; ++h) {
        conv3x3_tile_wmma<64, OUT_QK><<<dim3(4, 4, 256), 128, 0, stream>>>(
            aH, wqre + (size_t)h * 64 * 576, 64, nullptr, qh);
        conv3x3_tile_wmma<64, OUT_QK><<<dim3(4, 4, 256), 128, 0, stream>>>(
            aH, wkre + (size_t)h * 64 * 576, 64, nullptr, kh);
        conv3x3_tile_wmma<64, OUT_V><<<dim3(4, 4, 256), 128, 0, stream>>>(
            aH, wvre + (size_t)h * 64 * 576, 64, nullptr, vh);
        attn_s_wmma<<<dim3(4, 16), 128, 0, stream>>>(qh, kh, Smat);
        softmax256<<<256, 32, 0, stream>>>(Smat, Pmat);
        attn_pv_wmma<<<dim3(256, 16), 128, 0, stream>>>(Pmat, vh, oh, h);
      }
      // U projection: 512->64 per-tile 3x3 conv on attention output.
      conv3x3_tile_wmma<512, OUT_U><<<dim3(4, 4, 256), 128, 0, stream>>>(
          oh, wure, 64, tmp, nullptr);
      // identity = LN(u_out + x)
      residual_ln<<<256, 256, 0, stream>>>(tmp, aF, ln1w + (size_t)l * 16384,
                                           ln1b + (size_t)l * 16384, ident,
                                           identH, 0, nullptr);
      // MLP: 64->256 relu, 256->64 (+bias), then LN(mlp + identity) -> next t.
      gemm_wmma<64, GEMM_MLP1><<<dim3(1024, 16), 128, 0, stream>>>(
          w1re, identH, b1 + (size_t)l * 256, nullptr, m1);
      gemm_wmma<256, GEMM_MLP2><<<dim3(1024, 4), 128, 0, stream>>>(
          w2re, m1, b2 + (size_t)l * 64, tmp, nullptr);
      _Float16* img_b =
          (l == 1) ? (imgH + (size_t)b * PIMG * PIMG * 64) : nullptr;
      residual_ln<<<256, 256, 0, stream>>>(
          tmp, ident, ln2w + (size_t)l * 16384, ln2b + (size_t)l * 16384, aF,
          actH + (size_t)b * 256 * (PTILE * 64), 1, img_b);
    }
  }

  // Output head: full-image 3x3 conv 64->64 relu, then 1x1 conv 64->3.
  reorder_w_kernel<<<(64 * 576 + 255) / 256, 256, 0, stream>>>(ow1, wo1re, 64,
                                                               64, 9, 1.f);
  for (int b = 0; b < 2; ++b) {
    conv3x3_img_wmma<<<dim3(1024, 4), 128, 0, stream>>>(
        imgH + (size_t)b * PIMG * PIMG * 64, wo1re, ob1, h1H);
    out2_kernel<<<256, 256, 0, stream>>>(h1H, ow2, ob2, out, b);
  }
}
